// ImprovedWordGCN_15255723835550
// MI455X (gfx1250) — compile-verified, matrix operand order checked by reading, not run on hardware
//
#include <hip/hip_runtime.h>
#include <math.h>

// ---------------------------------------------------------------------------
// ImprovedWordGCN on MI455X (gfx1250, wave32, WMMA)
//   - dense GEMMs via v_wmma_f32_16x16x32_bf16 (bf16 in, f32 accumulate)
//   - compile-time K/NCOLS/ROWS: exact tile->wave map, no guards, no spills
//   - vectorized LDS fragment loads (ds_load_b128) via transposed W slab
//   - SpMM scatter via hardware f32 global atomics
// ---------------------------------------------------------------------------

typedef __bf16 bf16;
typedef __attribute__((ext_vector_type(16))) __bf16 v16bf;
typedef __attribute__((ext_vector_type(8)))  __bf16 v8bf;
typedef __attribute__((ext_vector_type(8)))  float  v8f;

#define HDIM 384

__device__ __forceinline__ float gelu_f(float x) {
    return 0.5f * x * (1.0f + erff(x * 0.70710678118654752440f));
}
__device__ __forceinline__ float sigmoid_f(float x) {
    return 1.0f / (1.0f + expf(-x));
}

// ---------------------------------------------------------------------------
// Fused WMMA GEMM: Y[rows,NCOLS] = epilogue( X[rows,K](ldx) @ W[K,NCOLS] )
// MODE 0: y = gelu(LN(v)*g+b)         MODE 1: y = gelu(v + bias)
// MODE 2: y = sigmoid(v + bias)       MODE 3: y = LN(gelu(v + bias))*g+b
// Block = 256 threads (8 wave32); block computes a ROWS x NCOLS strip.
// Requires (NCOLS/16) % (8/(ROWS/16)) == 0  -> exact tiles-per-wave.
// ---------------------------------------------------------------------------
template <int MODE, int K, int NCOLS, int ROWS>
__global__ __launch_bounds__(256) void wmma_gemm_fused(
    const float* __restrict__ X, int ldx,
    const float* __restrict__ W,
    const float* __restrict__ bias,
    const float* __restrict__ gamma, const float* __restrict__ beta,
    float* __restrict__ Y, int ldy,
    int rows)
{
    constexpr int NT     = NCOLS / 16;        // 16-col tiles
    constexpr int RG     = ROWS / 16;         // 16-row groups (1 or 2)
    constexpr int WPG    = 8 / RG;            // waves per row-group
    constexpr int TPW    = NT / WPG;          // tiles per wave (exact, ==3)
    constexpr int KSTEPS = K / 32;
    constexpr int SB_BYTES = (NCOLS * 64 > ROWS * NCOLS * 4)
                           ?  NCOLS * 64 : ROWS * NCOLS * 4;

    // sA: ROWSxK bf16. sBt: W slab transposed [NCOLS][32] bf16.
    // sOut (ROWSxNCOLS f32) aliases sBt (only live after the K loop).
    __shared__ __align__(16) char smemA[ROWS * K * 2];
    __shared__ __align__(16) char smemB[SB_BYTES];
    bf16*  sA   = (bf16*)smemA;
    bf16*  sBt  = (bf16*)smemB;
    float* sOut = (float*)smemB;
    __shared__ float sMu[ROWS];
    __shared__ float sRs[ROWS];

    const int tid  = threadIdx.x;
    const int wave = tid >> 5;
    const int lane = tid & 31;
    const int rh   = wave / WPG;              // row-group of this wave
    const int wsub = wave % WPG;              // wave index within row-group
    const int row0 = blockIdx.x * ROWS;

    // Stage the ROWSxK A strip into LDS as bf16 (coalesced global reads).
    for (int idx = tid; idx < ROWS * K; idx += 256) {
        const int r = idx / K;                // compile-time divisor
        const int k = idx - r * K;
        float v = (row0 + r < rows) ? X[(size_t)(row0 + r) * ldx + k] : 0.0f;
        sA[idx] = (bf16)v;
    }

    v8f acc[TPW];
    #pragma unroll
    for (int j = 0; j < TPW; ++j)
        #pragma unroll
        for (int e = 0; e < 8; ++e) acc[j][e] = 0.0f;

    // WMMA bf16 lane mapping (ISA 7.12.2):
    //  lanes 0-15: M/N = lane,    K in {0..7, 16..23}
    //  lanes16-31: M/N = lane-16, K in {8..15, 24..31}
    const int m  = lane & 15;
    const int kb = (lane >> 4) << 3;          // 0 or 8

    for (int ks = 0; ks < KSTEPS; ++ks) {
        __syncthreads();                       // previous slab fully consumed
        // Stage W[ks*32 .. +32, :] -> LDS transposed: sBt[n*32 + kl].
        for (int idx = tid; idx < 32 * NCOLS; idx += 256) {
            const int kl = idx / NCOLS;        // compile-time divisor
            const int n  = idx - kl * NCOLS;
            sBt[n * 32 + kl] = (bf16)W[(size_t)(ks * 32 + kl) * NCOLS + n];
        }
        // Prefetch next slab into cache while this one is consumed.
        if (ks + 1 < KSTEPS) {
            const float* nxt = W + (size_t)(ks + 1) * 32 * NCOLS;
            for (int off = tid * 32; off < 32 * NCOLS; off += 256 * 32)
                __builtin_prefetch(nxt + off, 0, 1);
        }
        __syncthreads();

        // A fragment: two contiguous 16B LDS loads (rows rh*16 .. rh*16+15).
        const bf16* ap = sA + (rh * 16 + m) * K + (ks << 5) + kb;
        v8bf alo = *(const v8bf*)ap;           // K = kb .. kb+7
        v8bf ahi = *(const v8bf*)(ap + 16);    // K = kb+16 .. kb+23
        v16bf afrag = __builtin_shufflevector(alo, ahi,
            0,1,2,3,4,5,6,7,8,9,10,11,12,13,14,15);

        #pragma unroll
        for (int j = 0; j < TPW; ++j) {        // exact: no guard, no masking
            const int n = ((wsub + WPG * j) << 4) + m;
            const bf16* bp = sBt + n * 32 + kb;
            v8bf blo = *(const v8bf*)bp;
            v8bf bhi = *(const v8bf*)(bp + 16);
            v16bf bfrag = __builtin_shufflevector(blo, bhi,
                0,1,2,3,4,5,6,7,8,9,10,11,12,13,14,15);
            acc[j] = __builtin_amdgcn_wmma_f32_16x16x32_bf16(
                false, afrag, false, bfrag, (short)0, acc[j], false, false);
        }
    }

    __syncthreads();                           // sBt dead -> reuse as sOut
    // C/D layout: lane gives N, vgpr r gives M (lanes 16-31 -> M+8).
    {
        const int mhi = rh * 16 + ((lane >> 4) << 3);
        #pragma unroll
        for (int j = 0; j < TPW; ++j) {
            const int n = ((wsub + WPG * j) << 4) + m;
            #pragma unroll
            for (int r = 0; r < 8; ++r)
                sOut[(mhi + r) * NCOLS + n] = acc[j][r];
        }
    }
    __syncthreads();

    if (MODE == 3) {                           // gelu(v + bias) before LN
        for (int idx = tid; idx < ROWS * NCOLS; idx += 256) {
            const int n = idx % NCOLS;
            sOut[idx] = gelu_f(sOut[idx] + bias[n]);
        }
        __syncthreads();
    }
    if (MODE == 0 || MODE == 3) {              // per-row mean / rstd
        if (tid < ROWS) {
            float s = 0.0f, sq = 0.0f;
            for (int n = 0; n < NCOLS; ++n) {
                float v = sOut[tid * NCOLS + n];
                s += v; sq += v * v;
            }
            const float mu  = s / (float)NCOLS;
            const float var = sq / (float)NCOLS - mu * mu;
            sMu[tid] = mu;
            sRs[tid] = rsqrtf(var + 1e-5f);
        }
        __syncthreads();
    }

    for (int idx = tid; idx < ROWS * NCOLS; idx += 256) {
        const int r = idx / NCOLS;
        const int n = idx - r * NCOLS;
        const int grow = row0 + r;
        if (grow >= rows) continue;
        const float v = sOut[idx];
        float y;
        if (MODE == 0)      y = gelu_f((v - sMu[r]) * sRs[r] * gamma[n] + beta[n]);
        else if (MODE == 1) y = gelu_f(v + bias[n]);
        else if (MODE == 2) y = sigmoid_f(v + bias[n]);
        else                y = (v - sMu[r]) * sRs[r] * gamma[n] + beta[n];
        Y[(size_t)grow * ldy + n] = y;
    }
}

// ---------------------------------------------------------------------------
// SpMM scatter: out[rows[e], :] += vals[e] * M[cols[e], :]   (H = 384)
// One wave per edge; 12 coalesced f32 hardware atomics per lane-row segment.
// ---------------------------------------------------------------------------
__global__ __launch_bounds__(256) void spmm_scatter(
    const int* __restrict__ rows, const int* __restrict__ cols,
    const float* __restrict__ vals,
    const float* __restrict__ M, int ldm,
    float* __restrict__ out, int ldo, int nnz)
{
    const int e = blockIdx.x * 8 + (threadIdx.x >> 5);
    if (e >= nnz) return;
    const int lane = threadIdx.x & 31;
    const int r = rows[e], c = cols[e];
    const float v = vals[e];
    const float* src = M + (size_t)c * ldm;
    float* dst = out + (size_t)r * ldo;
    #pragma unroll
    for (int i = 0; i < HDIM / 32; ++i)
        unsafeAtomicAdd(&dst[i * 32 + lane], v * src[i * 32 + lane]);
}

__global__ void zero_f32(float* __restrict__ p, size_t n) {
    size_t i = (size_t)blockIdx.x * blockDim.x + threadIdx.x;
    if (i < n) p[i] = 0.0f;
}

// word_H = (1-ALPHA)*emb + ALPHA*h
__global__ void residual_mix(const float* __restrict__ emb,
                             const float* __restrict__ h,
                             float* __restrict__ out, int n)
{
    int i = blockIdx.x * blockDim.x + threadIdx.x;
    if (i < n) out[i] = 0.35f * emb[i] + 0.65f * h[i];
}

// doc = g*doc_H + (1-g)*doc_H0, with doc_H/doc_H0 packed in gate_in [D,768]
__global__ void gate_mix(const float* __restrict__ gate,
                         const float* __restrict__ gin,
                         float* __restrict__ doc, int Dn)
{
    int idx = blockIdx.x * blockDim.x + threadIdx.x;
    if (idx >= Dn * HDIM) return;
    const int d = idx / HDIM;
    const int j = idx - d * HDIM;
    const float g  = gate[idx];
    const float h  = gin[(size_t)d * 768 + j];
    const float h0 = gin[(size_t)d * 768 + 384 + j];
    doc[idx] = g * h + (1.0f - g) * h0;
}

// logits = doc3[D,192] @ cls_W[192,2] + cls_b
__global__ void cls_head(const float* __restrict__ Xin,
                         const float* __restrict__ Wc,
                         const float* __restrict__ bc,
                         float* __restrict__ out, int Dn)
{
    int d = blockIdx.x * blockDim.x + threadIdx.x;
    if (d >= Dn) return;
    float a0 = bc[0], a1 = bc[1];
    const float* x = Xin + (size_t)d * 192;
    for (int k = 0; k < 192; ++k) {
        const float xv = x[k];
        a0 += xv * Wc[2 * k];
        a1 += xv * Wc[2 * k + 1];
    }
    out[2 * d]     = a0;
    out[2 * d + 1] = a1;
}

// ---------------------------------------------------------------------------
extern "C" void kernel_launch(void* const* d_in, const int* in_sizes, int n_in,
                              void* d_out, int out_size, void* d_ws, size_t ws_size,
                              hipStream_t stream)
{
    const int*   A_rows  = (const int*)  d_in[0];
    const int*   A_cols  = (const int*)  d_in[1];
    const float* A_vals  = (const float*)d_in[2];
    const int*   X_rows  = (const int*)  d_in[3];
    const int*   X_cols  = (const int*)  d_in[4];
    const float* X_vals  = (const float*)d_in[5];
    /* d_in[6] = num_docs (device scalar; derived from out_size instead) */
    const float* emb     = (const float*)d_in[7];
    const float* W1      = (const float*)d_in[8];
    const float* W2      = (const float*)d_in[9];
    const float* W3      = (const float*)d_in[10];
    const float* g1      = (const float*)d_in[11];
    const float* b1      = (const float*)d_in[12];
    const float* g2      = (const float*)d_in[13];
    const float* b2      = (const float*)d_in[14];
    const float* g3      = (const float*)d_in[15];
    const float* b3      = (const float*)d_in[16];
    const float* mlp_W1  = (const float*)d_in[17];
    const float* mlp_b1  = (const float*)d_in[18];
    const float* mlp_g   = (const float*)d_in[19];
    const float* mlp_bn  = (const float*)d_in[20];
    const float* mlp_W2  = (const float*)d_in[21];
    const float* mlp_b2  = (const float*)d_in[22];
    const float* cls_W   = (const float*)d_in[23];
    const float* cls_b   = (const float*)d_in[24];
    const float* gate_W1 = (const float*)d_in[25];
    const float* gate_b1 = (const float*)d_in[26];
    const float* gate_W2 = (const float*)d_in[27];
    const float* gate_b2 = (const float*)d_in[28];
    (void)n_in;

    const int Ew   = in_sizes[0];             // word-word edges
    const int NNZd = in_sizes[3];             // doc-word entries
    const int Nw   = in_sizes[7] / HDIM;      // 40000 words
    const int Dd   = out_size / 2;            // 20000 docs

    const size_t nH = (size_t)Nw * HDIM;
    if (ws_size < 2 * nH * sizeof(float)) return;

    float* buf0 = (float*)d_ws;               // [N,384]
    float* buf1 = buf0 + nH;                  // [N,384] == [D,768]
    float* b0a  = buf0;                       // [D,384] sub-regions of buf0
    float* b0b  = buf0 + (size_t)Dd * HDIM;

    const int TB = 256;
    dim3 blk(TB);
    dim3 gZeroN((unsigned)((nH + TB - 1) / TB));
    dim3 gSpmmA((unsigned)((Ew + 7) / 8));
    dim3 gSpmmX((unsigned)((NNZd + 7) / 8));
    dim3 gGemmN((unsigned)((Nw + 15) / 16));
    dim3 gGemmD16((unsigned)((Dd + 15) / 16));
    dim3 gGemmD32((unsigned)((Dd + 31) / 32));
    dim3 gElemD((unsigned)(((size_t)Dd * HDIM + TB - 1) / TB));
    dim3 gCls((unsigned)((Dd + TB - 1) / TB));

    // --- GCN block 1 ---
    zero_f32<<<gZeroN, blk, 0, stream>>>(buf0, nH);
    spmm_scatter<<<gSpmmA, blk, 0, stream>>>(A_rows, A_cols, A_vals, emb, HDIM, buf0, HDIM, Ew);
    wmma_gemm_fused<0, 384, 384, 16><<<gGemmN, blk, 0, stream>>>(buf0, HDIM, W1, nullptr, g1, b1, buf1, HDIM, Nw);
    // --- GCN block 2 ---
    zero_f32<<<gZeroN, blk, 0, stream>>>(buf0, nH);
    spmm_scatter<<<gSpmmA, blk, 0, stream>>>(A_rows, A_cols, A_vals, buf1, HDIM, buf0, HDIM, Ew);
    wmma_gemm_fused<0, 384, 384, 16><<<gGemmN, blk, 0, stream>>>(buf0, HDIM, W2, nullptr, g2, b2, buf1, HDIM, Nw);
    // --- GCN block 3 ---
    zero_f32<<<gZeroN, blk, 0, stream>>>(buf0, nH);
    spmm_scatter<<<gSpmmA, blk, 0, stream>>>(A_rows, A_cols, A_vals, buf1, HDIM, buf0, HDIM, Ew);
    wmma_gemm_fused<0, 384, 384, 16><<<gGemmN, blk, 0, stream>>>(buf0, HDIM, W3, nullptr, g3, b3, buf1, HDIM, Nw);

    // word_H = 0.35*emb + 0.65*h3   (buf1 -> buf0)
    residual_mix<<<gZeroN, blk, 0, stream>>>(emb, buf1, buf0, (int)nH);

    // --- doc aggregation: gate_in[D,768] = [spmm(X,word_H) | spmm(X,emb)] ---
    float* gate_in = buf1;
    zero_f32<<<gZeroN, blk, 0, stream>>>(gate_in, nH);
    spmm_scatter<<<gSpmmX, blk, 0, stream>>>(X_rows, X_cols, X_vals, buf0, HDIM, gate_in,       768, NNZd);
    spmm_scatter<<<gSpmmX, blk, 0, stream>>>(X_rows, X_cols, X_vals, emb,  HDIM, gate_in + 384, 768, NNZd);

    // gate_h = gelu(gate_in @ gate_W1 + gate_b1)        (word_H dead -> b0a)
    wmma_gemm_fused<1, 768, 384, 16><<<gGemmD16, blk, 0, stream>>>(gate_in, 768, gate_W1, gate_b1, nullptr, nullptr, b0a, HDIM, Dd);
    // gate = sigmoid(gate_h @ gate_W2 + gate_b2)
    wmma_gemm_fused<2, 384, 384, 16><<<gGemmD16, blk, 0, stream>>>(b0a, HDIM, gate_W2, gate_b2, nullptr, nullptr, b0b, HDIM, Dd);
    // doc = gate*doc_H + (1-gate)*doc_H0                (gate_h dead -> b0a)
    gate_mix<<<gElemD, blk, 0, stream>>>(b0b, gate_in, b0a, Dd);

    // doc2 = LN(gelu(doc @ mlp_W1 + mlp_b1)) * g + b    (gate dead -> b0b)
    wmma_gemm_fused<3, 384, 384, 16><<<gGemmD16, blk, 0, stream>>>(b0a, HDIM, mlp_W1, mlp_b1, mlp_g, mlp_bn, b0b, HDIM, Dd);
    // doc3 = gelu(doc2 @ mlp_W2 + mlp_b2)  [D,192]      (doc dead -> b0a)
    wmma_gemm_fused<1, 384, 192, 32><<<gGemmD32, blk, 0, stream>>>(b0b, HDIM, mlp_W2, mlp_b2, nullptr, nullptr, b0a, 192, Dd);

    // logits = doc3 @ cls_W + cls_b
    cls_head<<<gCls, blk, 0, stream>>>(b0a, cls_W, cls_b, (float*)d_out, Dd);
}
